// DAModule_377957122715
// MI455X (gfx1250) — matrix operand order repaired and checked by reading, not compiled, not run on hardware
//
#include <hip/hip_runtime.h>
#include <hip/hip_bf16.h>

// ---------------------------------------------------------------------------
// DAModule for MI455X / gfx1250.
// Conv = implicit GEMM on v_wmma_f32_16x16x32_f16, wave32.
// Activations: NHWC f16 (C padded to multiple of 32) so each B fragment is a
// single aligned 32B contiguous load. Weights pre-packed into per-lane
// wave32 A fragments (k = 16*half + s, as B uses); A fragments are staged into LDS once per
// block (two tap-phases, <=40KB) and read via ds_load in the hot loop.
// Each wave: 16 pixels x 64 out-channels => 4 WMMA accumulators per B load.
// ---------------------------------------------------------------------------

typedef __attribute__((ext_vector_type(16))) _Float16 v16h;
typedef __attribute__((ext_vector_type(8)))  _Float16 v8h;
typedef __attribute__((ext_vector_type(8)))  float    v8f;

__device__ __forceinline__ float lrelu_f(float v) { return v >= 0.0f ? v : 0.2f * v; }

// ---------------------------------------------------------------------------
// Implicit-GEMM conv.  K ordered k = tap*Cp + c  (Cp multiple of 32).
//   x     : NHWC f16, channels = Cp
//   apack : A fragments, layout [n][kc][mt][lane] of v16h (from pack_a_k)
//   yh    : NHWC f16 output (out_f32==0)
//   yf    : NCHW f32 output (out_f32==1)
// Block = 256 threads = 8 waves; each wave owns 16 pixels x MTtot*16 outs.
// Dynamic LDS: 5 * (Cp/32) * MTtot * 32 * sizeof(v16h) bytes (max 40KB).
// ---------------------------------------------------------------------------
__global__ __launch_bounds__(256)
void conv_wmma_k(const _Float16* __restrict__ x, const v16h* __restrict__ apack,
                 const float* __restrict__ bias,
                 _Float16* __restrict__ yh, float* __restrict__ yf,
                 int Cp, int OUT, int Hin, int Win, int Hout, int Wout,
                 int stride, int pad, int ldil, int bns, int act,
                 int MTtot, int KC, int out_f32)
{
    extern __shared__ __align__(32) char smem_raw[];
    v16h* sA = (v16h*)smem_raw;

    const int lane = threadIdx.x & 31;
    const int wave = threadIdx.x >> 5;
    const int n    = blockIdx.z;
    const int P    = Hout * Wout;
    const int pix0 = (blockIdx.x * 8 + wave) * 16;

    const int l15  = lane & 15;
    const int half = lane >> 4;
    const int p    = pix0 + l15;
    const bool pvalid = (p < P);
    const int oy = pvalid ? (p / Wout) : 0;
    const int ox = pvalid ? (p % Wout) : 0;

    const int CB = Cp >> 5;                 // chunks per tap (1 or 2)
    const _Float16* __restrict__ xb = x + (size_t)n * Hin * Win * Cp;
    const v16h* __restrict__ apn = apack + (size_t)n * KC * MTtot * 32;

    v8f acc0 = {}, acc1 = {}, acc2 = {}, acc3 = {};

    #pragma unroll
    for (int ph = 0; ph < 2; ++ph) {
        const int tap0 = ph ? 5 : 0;
        const int tapN = ph ? 9 : 5;

        // ---- stage this phase's A fragments into LDS ----------------------
        __syncthreads();                     // protect prior phase's reads
        {
            const int cnt = (tapN - tap0) * CB * MTtot * 32;
            const size_t src0 = (size_t)tap0 * CB * MTtot * 32;
            for (int i = threadIdx.x; i < cnt; i += 256)
                sA[i] = apn[src0 + i];
        }
        __syncthreads();

        int kcl = 0;                         // chunk index within phase
        #pragma unroll
        for (int tap = tap0; tap < tapN; ++tap) {
            const int ky = tap / 3, kx = tap % 3;   // constants after unroll
            const int ty = oy * stride + ky - pad;
            const int tx = ox * stride + kx - pad;
            bool ok = pvalid && ty >= 0 && tx >= 0;
            int iy, ix;
            if (ldil == 2) {                 // transposed conv: zero-insert
                ok = ok && ((ty & 1) == 0) && ((tx & 1) == 0);
                iy = ty >> 1; ix = tx >> 1;
            } else { iy = ty; ix = tx; }
            ok = ok && iy < Hin && ix < Win;
            const long long bbase = ((long long)iy * Win + ix) * Cp + half * 16;

            for (int cb = 0; cb < Cp; cb += 32) {
                v16h bf = {};
                if (ok) bf = *(const v16h*)(xb + bbase + cb);   // one 32B load

                const v16h* aptr = sA + (size_t)kcl * MTtot * 32 + lane;
                acc0 = __builtin_amdgcn_wmma_f32_16x16x32_f16(
                           false, aptr[0], false, bf, (short)0, acc0, false, false);
                if (MTtot > 1)
                    acc1 = __builtin_amdgcn_wmma_f32_16x16x32_f16(
                               false, aptr[32], false, bf, (short)0, acc1, false, false);
                if (MTtot > 2)
                    acc2 = __builtin_amdgcn_wmma_f32_16x16x32_f16(
                               false, aptr[64], false, bf, (short)0, acc2, false, false);
                if (MTtot > 3)
                    acc3 = __builtin_amdgcn_wmma_f32_16x16x32_f16(
                               false, aptr[96], false, bf, (short)0, acc3, false, false);
                ++kcl;
            }
        }
    }

    v8f accs[4] = {acc0, acc1, acc2, acc3};

    if (!out_f32) {
        if (pvalid) {
            _Float16* yp = yh + (size_t)p * OUT;
            #pragma unroll
            for (int mt = 0; mt < 4; ++mt) {
                if (mt < MTtot) {                    // uniform branch
                    v8h o;
                    #pragma unroll
                    for (int r = 0; r < 8; ++r) {
                        float v = accs[mt][r];
                        if (bias) v += bias[(size_t)n * bns + mt * 16 + 8 * half + r];
                        if (act)  v = lrelu_f(v);
                        o[r] = (_Float16)v;
                    }
                    *(v8h*)(yp + mt * 16 + 8 * half) = o;   // one 16B store
                }
            }
        }
    } else {
        #pragma unroll
        for (int mt = 0; mt < 4; ++mt) {
            if (mt < MTtot) {
                #pragma unroll
                for (int r = 0; r < 8; ++r) {
                    const int mo = mt * 16 + 8 * half + r;
                    if (mo < OUT && pvalid) {
                        float v = accs[mt][r];
                        if (bias) v += bias[(size_t)n * bns + mo];
                        if (act)  v = lrelu_f(v);
                        yf[((size_t)n * OUT + mo) * (size_t)P + p] = v;
                    }
                }
            }
        }
    }
}

// ---------------------------------------------------------------------------
// Pack f32 weights [.. , OUT, Creal, 3,3] into A-fragment layout f16.
// One thread per (n, kc, mt, lane): fills 16 halves (32B vector store).
// k = tap*Cp + c ;  Cp = 1<<cshift ;  channels >= Creal and k >= 9*Cp are 0.
// flip: tap -> 8-tap (transposed conv).  wns: per-sample weight stride.
// ---------------------------------------------------------------------------
__global__ __launch_bounds__(64)
void pack_a_k(const float* __restrict__ w, v16h* __restrict__ apack,
              int wns, int OUT, int Creal, int cshift, int MT, int KC,
              int flip, int total)
{
    const int i = blockIdx.x * 64 + threadIdx.x;   // ((n*KC+kc)*MT+mt)*32+lane
    if (i >= total) return;
    const int lane = i & 31;
    int t = i >> 5;
    const int mt = t % MT; t /= MT;
    const int kc = t % KC;
    const int n  = t / KC;
    const int l15 = lane & 15, half = lane >> 4;
    const int m = mt * 16 + l15;
    const int cmask = (1 << cshift) - 1;
    const float* wb = w + (size_t)n * wns;

    v16h a;
    #pragma unroll
    for (int e = 0; e < 16; ++e) {
        const int kb = half * 16 + e;
        const int k = kc * 32 + kb;
        const int tap = k >> cshift;
        const int c = k & cmask;
        float val = 0.0f;
        if (m < OUT && c < Creal && tap < 9)
            val = wb[((size_t)m * Creal + c) * 9 + (flip ? (8 - tap) : tap)];
        a[e] = (_Float16)val;
    }
    apack[i] = a;
}

// ---------------------------------------------------------------------------
// NCHW f32 -> NHWC f16 with channel zero-padding to Cp.
// ---------------------------------------------------------------------------
__global__ __launch_bounds__(256)
void to_nhwc_k(const float* __restrict__ x, _Float16* __restrict__ y,
               int Cin, int Cp, int HW, size_t total)
{
    const size_t idx = (size_t)blockIdx.x * 256 + threadIdx.x;
    if (idx >= total) return;
    const int c = (int)(idx % Cp);
    const size_t rest = idx / Cp;
    const int pix = (int)(rest % HW);
    const int n = (int)(rest / HW);
    y[idx] = (c < Cin) ? (_Float16)x[((size_t)n * Cin + c) * HW + pix] : (_Float16)0.0f;
}

// ---------------------------------------------------------------------------
// s[n,c] = mean over HW of NHWC f16 (block per (n,c))
// ---------------------------------------------------------------------------
__global__ __launch_bounds__(256)
void mean_nhwc_k(const _Float16* __restrict__ x, float* __restrict__ out,
                 int HW, int C)
{
    __shared__ float red[256];
    const int n = blockIdx.x / C, c = blockIdx.x % C;
    const _Float16* p = x + (size_t)n * HW * C + c;
    float sum = 0.0f;
    for (int i = threadIdx.x; i < HW; i += 256) sum += (float)p[(size_t)i * C];
    red[threadIdx.x] = sum;
    __syncthreads();
    for (int s = 128; s > 0; s >>= 1) {
        if ((int)threadIdx.x < s) red[threadIdx.x] += red[threadIdx.x + s];
        __syncthreads();
    }
    if (threadIdx.x == 0) out[blockIdx.x] = red[0] / (float)HW;
}

// ---------------------------------------------------------------------------
// out[n,i] = lrelu( dot(s[n,:], W[i,:]) + b[i] ),  W is [OD, F]
// ---------------------------------------------------------------------------
__global__ __launch_bounds__(64)
void style_k(const float* __restrict__ s, const float* __restrict__ Wt,
             const float* __restrict__ b, float* __restrict__ out,
             int OD, int F, int total)
{
    const int i = blockIdx.x * 64 + threadIdx.x;
    if (i >= total) return;
    const int n = i / OD;
    const int o = i - n * OD;
    const float* sv = s + (size_t)n * F;
    const float* wv = Wt + (size_t)o * F;
    float acc = b[o];
    for (int j = 0; j < F; ++j) acc += sv[j] * wv[j];
    out[i] = lrelu_f(acc);
}

// ---------------------------------------------------------------------------
// Modulate + demodulate (one block per (n,o)):
//   t = w[o,c,tap]*sw[n,c]; d = rsqrt(sum t^2 + eps); wout[n,o,c,tap] = t*d
// ---------------------------------------------------------------------------
__global__ __launch_bounds__(64)
void modulate_k(const float* __restrict__ w, const float* __restrict__ sw,
                float* __restrict__ wout, int C, int OUT)
{
    const int n = blockIdx.x / OUT;
    const int o = blockIdx.x - n * OUT;
    const int K = C * 9;
    __shared__ float red[64];
    float ss = 0.0f;
    for (int k = threadIdx.x; k < K; k += 64) {
        const int c = k / 9;
        const float v = w[(size_t)o * K + k] * sw[(size_t)n * C + c];
        ss += v * v;
    }
    red[threadIdx.x] = ss;
    __syncthreads();
    for (int st = 32; st > 0; st >>= 1) {
        if ((int)threadIdx.x < st) red[threadIdx.x] += red[threadIdx.x + st];
        __syncthreads();
    }
    const float d = rsqrtf(red[0] + 1e-8f);
    for (int k = threadIdx.x; k < K; k += 64) {
        const int c = k / 9;
        wout[((size_t)n * OUT + o) * K + k] =
            w[(size_t)o * K + k] * sw[(size_t)n * C + c] * d;
    }
}

// ---------------------------------------------------------------------------
// Depthwise 4x4 FIR blur on NHWC f16. Hout = Hin + 2*pad - 3.
// ---------------------------------------------------------------------------
__global__ __launch_bounds__(256)
void blur4_k(const _Float16* __restrict__ x, _Float16* __restrict__ y,
             int C, int Hin, int Win, int pad, float scale, size_t total)
{
    const size_t idx = (size_t)blockIdx.x * 256 + threadIdx.x;
    if (idx >= total) return;
    const int Hout = Hin + 2 * pad - 3;
    const int Wout = Win + 2 * pad - 3;
    const int c = (int)(idx % C);
    size_t rest = idx / C;
    const int ox = (int)(rest % Wout); rest /= Wout;
    const int oy = (int)(rest % Hout);
    const int n = (int)(rest / Hout);

    const float kv[4] = {1.0f, 3.0f, 3.0f, 1.0f};
    const _Float16* xp = x + (size_t)n * Hin * Win * C + c;
    float acc = 0.0f;
    #pragma unroll
    for (int i = 0; i < 4; ++i) {
        const int iy = oy + i - pad;
        if (iy < 0 || iy >= Hin) continue;
        #pragma unroll
        for (int j = 0; j < 4; ++j) {
            const int ix = ox + j - pad;
            if (ix < 0 || ix >= Win) continue;
            acc += kv[i] * kv[j] * (float)xp[((size_t)iy * Win + ix) * C];
        }
    }
    y[idx] = (_Float16)(acc * (scale / 64.0f));
}

// ---------------------------------------------------------------------------
// In-place y = lrelu(y + sb[n,c]) on NHWC f16
// ---------------------------------------------------------------------------
__global__ __launch_bounds__(256)
void bias_act_k(_Float16* __restrict__ y, const float* __restrict__ sb,
                int C, int HW, size_t total)
{
    const size_t idx = (size_t)blockIdx.x * 256 + threadIdx.x;
    if (idx >= total) return;
    const int c = (int)(idx % C);
    const int n = (int)(idx / ((size_t)HW * C));
    y[idx] = (_Float16)lrelu_f((float)y[idx] + sb[(size_t)n * C + c]);
}

__global__ __launch_bounds__(256)
void add_k(const _Float16* __restrict__ a, const _Float16* __restrict__ b,
           _Float16* __restrict__ c, size_t total)
{
    const size_t idx = (size_t)blockIdx.x * 256 + threadIdx.x;
    if (idx < total) c[idx] = (_Float16)((float)a[idx] + (float)b[idx]);
}

// ---------------------------------------------------------------------------
// Host orchestration
// ---------------------------------------------------------------------------
extern "C" void kernel_launch(void* const* d_in, const int* in_sizes, int n_in,
                              void* d_out, int out_size, void* d_ws, size_t ws_size,
                              hipStream_t stream)
{
    (void)in_sizes; (void)n_in; (void)out_size; (void)ws_size;
    constexpr int N = 8, F = 64, INC = 3, OUTC = 3, H = 256, W = 256;

    const float* x      = (const float*)d_in[0];
    const float* ref    = (const float*)d_in[1];
    const float* g0_w   = (const float*)d_in[2];
    const float* g0_b   = (const float*)d_in[3];
    const float* gr_w   = (const float*)d_in[4];
    const float* gr_b   = (const float*)d_in[5];
    const float* c0_w   = (const float*)d_in[6];
    const float* c0_swW = (const float*)d_in[7];
    const float* c0_swb = (const float*)d_in[8];
    const float* c0_sbW = (const float*)d_in[9];
    const float* c0_sbb = (const float*)d_in[10];
    const float* mid_w   = (const float*)d_in[11];
    const float* mid_swW = (const float*)d_in[12];
    const float* mid_swb = (const float*)d_in[13];
    const float* mid_sbW = (const float*)d_in[14];
    const float* mid_sbb = (const float*)d_in[15];
    const float* c6_w   = (const float*)d_in[16];
    const float* c6_swW = (const float*)d_in[17];
    const float* c6_swb = (const float*)d_in[18];
    const float* c6_sbW = (const float*)d_in[19];
    const float* c6_sbb = (const float*)d_in[20];
    float* out = (float*)d_out;

    // ---- workspace arena (byte carve, 256B aligned) -----------------------
    char* base = (char*)d_ws;
    size_t off = 0;
    auto carve = [&](size_t bytes) {
        off = (off + 255) & ~(size_t)255;
        void* p = base + off; off += bytes; return p;
    };
    float*    s_buf  = (float*)carve((size_t)N * F * 4);
    float*    sw_buf = (float*)carve((size_t)N * F * 4);
    float*    sb_buf = (float*)carve((size_t)N * F * 4);
    float*    wmod   = (float*)carve((size_t)N * F * F * 9 * 4);
    v16h*     apack  = (v16h*)carve((size_t)N * 18 * 4 * 32 * sizeof(v16h));
    _Float16* xh     = (_Float16*)carve((size_t)N * H * W * 32 * 2);
    _Float16* refh   = (_Float16*)carve((size_t)N * H * W * 32 * 2);
    const size_t big = (size_t)N * 257 * 257 * F * 2;
    _Float16* bufA   = (_Float16*)carve(big);
    _Float16* bufB   = (_Float16*)carve(big);
    _Float16* bufC   = (_Float16*)carve(big);
    _Float16* f0k    = (_Float16*)carve((size_t)N * 256 * 256 * F * 2);
    _Float16* f1k    = (_Float16*)carve((size_t)N * 128 * 128 * F * 2);
    _Float16* f2k    = (_Float16*)carve((size_t)N * 64 * 64 * F * 2);

    // ---- launch helpers ---------------------------------------------------
    auto conv = [&](const _Float16* xi, const float* bi, int bns,
                    _Float16* yh, float* yf,
                    int Cp, int OUT, int Hin, int Win, int Hout, int Wout,
                    int stride, int pad, int ldil, int act) {
        const int P = Hout * Wout;
        const int MT = (OUT + 15) / 16;
        const int KC = 9 * (Cp >> 5);
        const size_t shb = (size_t)5 * (Cp >> 5) * MT * 32 * sizeof(v16h);
        dim3 g((unsigned)((P + 127) / 128), 1, N);
        conv_wmma_k<<<g, 256, shb, stream>>>(xi, apack, bi, yh, yf, Cp, OUT,
                                             Hin, Win, Hout, Wout, stride, pad,
                                             ldil, bns, act, MT, KC, yf != nullptr);
    };
    auto pack = [&](const float* wsrc, int wns, int OUT, int Creal,
                    int cshift, int flip) {
        const int MT = (OUT + 15) / 16;
        const int KC = 9 << (cshift - 5);
        const int total = N * KC * MT * 32;
        pack_a_k<<<(total + 63) / 64, 64, 0, stream>>>(wsrc, apack, wns, OUT,
                                                       Creal, cshift, MT, KC,
                                                       flip, total);
    };
    auto style = [&](const float* Wt, const float* b, float* o, int OD) {
        const int tot = N * OD;
        style_k<<<(tot + 63) / 64, 64, 0, stream>>>(s_buf, Wt, b, o, OD, F, tot);
    };
    auto modulate = [&](const float* wi, int C, int OUT) {
        modulate_k<<<N * OUT, 64, 0, stream>>>(wi, sw_buf, wmod, C, OUT);
    };
    auto blur = [&](const _Float16* xi, _Float16* yo, int Hin, int Win,
                    int pad, float sc) {
        const int Ho = Hin + 2 * pad - 3, Wo = Win + 2 * pad - 3;
        const size_t tot = (size_t)N * Ho * Wo * F;
        blur4_k<<<(unsigned)((tot + 255) / 256), 256, 0, stream>>>(
            xi, yo, F, Hin, Win, pad, sc, tot);
    };
    auto biasact = [&](_Float16* yo, int HW) {
        const size_t tot = (size_t)N * HW * F;
        bias_act_k<<<(unsigned)((tot + 255) / 256), 256, 0, stream>>>(
            yo, sb_buf, F, HW, tot);
    };
    auto add = [&](const _Float16* a, const _Float16* b, _Float16* c, size_t tot) {
        add_k<<<(unsigned)((tot + 255) / 256), 256, 0, stream>>>(a, b, c, tot);
    };

    const int WNS = F * F * 9;

    // ===== guide net (shared weights => wns=0, bns=0) ======================
    {
        const size_t tot = (size_t)N * H * W * 32;
        to_nhwc_k<<<(unsigned)((tot + 255) / 256), 256, 0, stream>>>(
            ref, refh, INC, 32, H * W, tot);
    }
    pack(g0_w, 0, F, INC, 5, 0);
    conv(refh, g0_b, 0, bufA, nullptr, 32, F, 256,256, 256,256, 1,1,1, 1);
    pack(gr_w + 0 * WNS, 0, F, F, 6, 0);
    conv(bufA, gr_b + 0 * F, 0, bufB, nullptr, F, F, 256,256, 128,128, 2,1,1, 1);
    pack(gr_w + 1 * WNS, 0, F, F, 6, 0);
    conv(bufB, gr_b + 1 * F, 0, bufA, nullptr, F, F, 128,128, 128,128, 1,1,1, 1);
    pack(gr_w + 2 * WNS, 0, F, F, 6, 0);
    conv(bufA, gr_b + 2 * F, 0, bufB, nullptr, F, F, 128,128, 64,64, 2,1,1, 1);
    pack(gr_w + 3 * WNS, 0, F, F, 6, 0);
    conv(bufB, gr_b + 3 * F, 0, bufA, nullptr, F, F, 64,64, 64,64, 1,1,1, 0);
    mean_nhwc_k<<<N * F, 256, 0, stream>>>(bufA, s_buf, 64 * 64, F);

    auto mid_style = [&](int i, int flip) {
        style(mid_swW + (size_t)i * F * F, mid_swb + (size_t)i * F, sw_buf, F);
        style(mid_sbW + (size_t)i * F * F, mid_sbb + (size_t)i * F, sb_buf, F);
        modulate(mid_w + (size_t)i * F * F * 9, F, F);
        pack(wmod, WNS, F, F, 6, flip);
    };

    // ===== c0: 'none' 3->64 @256 ==========================================
    {
        const size_t tot = (size_t)N * H * W * 32;
        to_nhwc_k<<<(unsigned)((tot + 255) / 256), 256, 0, stream>>>(
            x, xh, INC, 32, H * W, tot);
    }
    style(c0_swW, c0_swb, sw_buf, INC);
    style(c0_sbW, c0_sbb, sb_buf, F);
    modulate(c0_w, INC, F);
    pack(wmod, F * INC * 9, F, INC, 5, 0);
    conv(xh, sb_buf, F, f0k, nullptr, 32, F, 256,256, 256,256, 1,1,1, 1);

    // ===== mid0 'down' 256->128 ===========================================
    mid_style(0, 0);
    blur(f0k, bufA, 256, 256, 2, 1.0f);                     // -> 257x257
    conv(bufA, sb_buf, F, bufB, nullptr, F, F, 257,257, 128,128, 2,0,1, 1);
    // ===== mid1 'none' @128 -> f1 =========================================
    mid_style(1, 0);
    conv(bufB, sb_buf, F, f1k, nullptr, F, F, 128,128, 128,128, 1,1,1, 1);
    // ===== mid2 'down' 128->64 ============================================
    mid_style(2, 0);
    blur(f1k, bufA, 128, 128, 2, 1.0f);                     // -> 129x129
    conv(bufA, sb_buf, F, bufB, nullptr, F, F, 129,129, 64,64, 2,0,1, 1);
    // ===== mid3 'none' @64 -> f2 ==========================================
    mid_style(3, 0);
    conv(bufB, sb_buf, F, f2k, nullptr, F, F, 64,64, 64,64, 1,1,1, 1);
    // ===== mid4, mid5 'none' @64 ==========================================
    mid_style(4, 0);
    conv(f2k, sb_buf, F, bufB, nullptr, F, F, 64,64, 64,64, 1,1,1, 1);
    mid_style(5, 0);
    conv(bufB, sb_buf, F, bufA, nullptr, F, F, 64,64, 64,64, 1,1,1, 1);
    add(bufA, f2k, bufB, (size_t)N * F * 64 * 64);          // f3 + f2
    // ===== mid6 'up' 64->128 ==============================================
    mid_style(6, 1);                                        // flipped kernel
    conv(bufB, nullptr, 0, bufC, nullptr, F, F, 64,64, 129,129, 1,2,2, 0);
    blur(bufC, bufA, 129, 129, 1, 4.0f);                    // -> 128x128
    biasact(bufA, 128 * 128);
    // ===== mid7 'none' @128 ===============================================
    mid_style(7, 0);
    conv(bufA, sb_buf, F, bufB, nullptr, F, F, 128,128, 128,128, 1,1,1, 1);
    add(bufB, f1k, bufA, (size_t)N * F * 128 * 128);        // f4 + f1
    // ===== mid8 'up' 128->256 =============================================
    mid_style(8, 1);
    conv(bufA, nullptr, 0, bufC, nullptr, F, F, 128,128, 257,257, 1,2,2, 0);
    blur(bufC, bufB, 257, 257, 1, 4.0f);                    // -> 256x256
    biasact(bufB, 256 * 256);
    // ===== mid9 'none' @256 ===============================================
    mid_style(9, 0);
    conv(bufB, sb_buf, F, bufA, nullptr, F, F, 256,256, 256,256, 1,1,1, 1);
    add(bufA, f0k, bufB, (size_t)N * F * 256 * 256);        // f5 + f0
    // ===== c6 'none' 64->3 @256, f32 NCHW out, no act =====================
    style(c6_swW, c6_swb, sw_buf, F);
    style(c6_sbW, c6_sbb, sb_buf, OUTC);
    modulate(c6_w, F, OUTC);
    pack(wmod, OUTC * F * 9, OUTC, F, 6, 0);
    conv(bufB, sb_buf, OUTC, nullptr, out, F, OUTC, 256,256, 256,256, 1,1,1, 0);
}
